// MultiHeadAttention_88261577932867
// MI455X (gfx1250) — matrix operand order repaired and checked by reading, not compiled, not run on hardware
//
#include <hip/hip_runtime.h>
#include <hip/hip_bf16.h>

typedef __attribute__((ext_vector_type(16))) __bf16 v16bf;
typedef __attribute__((ext_vector_type(8)))  __bf16 v8bf;
typedef __attribute__((ext_vector_type(8)))  float  v8f;

namespace {
constexpr int kB  = 2;
constexpr int kS  = 2048;
constexpr int kE  = 1024;
constexpr int kH  = 16;
constexpr int kD  = 64;
constexpr int k3E = 3 * kE;

__device__ __forceinline__ unsigned short f32_to_bf16_bits(float f) {
  unsigned int u = __float_as_uint(f);
  u += 0x7FFFu + ((u >> 16) & 1u);   // round-to-nearest-even
  return (unsigned short)(u >> 16);
}
__device__ __forceinline__ __bf16 f32_to_bf16(float f) {
  unsigned short s = f32_to_bf16_bits(f);
  return __builtin_bit_cast(__bf16, s);
}

// D = A(16x32 bf16) * B(32x16 bf16) + C(16x16 f32)
__device__ __forceinline__ v8f wmma_bf16(v16bf a, v16bf b, v8f c) {
  return __builtin_amdgcn_wmma_f32_16x16x32_bf16(false, a, false, b, (short)0, c,
                                                 false, false);
}

__device__ __forceinline__ v16bf concat8(v8bf lo, v8bf hi) {
  return __builtin_shufflevector(lo, hi, 0, 1, 2, 3, 4, 5, 6, 7,
                                         8, 9, 10, 11, 12, 13, 14, 15);
}

// A-fragment 16x32 from bf16 row-major source (base -> element (0,0)).
// Per lane: row m=lane&15; K-runs [16*half, +8) and [16*half+8, +8)
// => two contiguous 16B loads (global_load_b128 / ds_load_b128).
__device__ __forceinline__ v16bf load_A_bf16(const __bf16* base, int ld, int lane) {
  const int half = lane >> 4;
  const __bf16* p = base + (size_t)(lane & 15) * ld + half * 16;
  v8bf lo = *(const v8bf*)(p);
  v8bf hi = *(const v8bf*)(p + 8);
  return concat8(lo, hi);
}

// B-fragment 32x16 where element (k,n) = base[n*ld + k] (N-major source).
// Per lane: 16 contiguous bf16 starting at k = 16*half => one 32B vector load.
__device__ __forceinline__ v16bf load_B_nmajor(const __bf16* base, int ld, int lane) {
  const __bf16* p = base + (size_t)(lane & 15) * ld + (lane >> 4) * 16;
  return *(const v16bf*)p;
}
} // namespace

// ---------------- Kernel 0a: x f32 -> bf16 (row-major) ----------------
__global__ __launch_bounds__(256) void cvt_f32_bf16(const float* __restrict__ in,
                                                    __bf16* __restrict__ out, int nel) {
  int i = blockIdx.x * 256 + threadIdx.x;
  if (i < nel) out[i] = f32_to_bf16(in[i]);
}

// ---------------- Kernel 0b: transpose f32[R,C] -> bf16[C,R] ----------------
__global__ __launch_bounds__(256) void transpose_f32_bf16(const float* __restrict__ in,
                                                          __bf16* __restrict__ out,
                                                          int R, int C) {
  __shared__ float tile[32][33];
  const int c0 = blockIdx.x * 32, r0 = blockIdx.y * 32;
  const int tx = threadIdx.x & 31, ty = threadIdx.x >> 5;  // 32 x 8
#pragma unroll
  for (int i = 0; i < 32; i += 8)
    tile[ty + i][tx] = in[(size_t)(r0 + ty + i) * C + c0 + tx];
  __syncthreads();
#pragma unroll
  for (int i = 0; i < 32; i += 8)
    out[(size_t)(c0 + ty + i) * R + r0 + tx] = f32_to_bf16(tile[tx][ty + i]);
}

// Ping-pong GEMM macros: load one buffer while computing with the other.
#define GEMM_LOAD(A0, A1, B0, B1, B2, B3, APTR, BPTR, KK)                        \
  A0 = load_A_bf16((APTR) + (KK), kE, lane);                                     \
  A1 = load_A_bf16((APTR) + (size_t)16 * kE + (KK), kE, lane);                   \
  B0 = load_B_nmajor((BPTR) + (KK), kE, lane);                                   \
  B1 = load_B_nmajor((BPTR) + (size_t)16 * kE + (KK), kE, lane);                 \
  B2 = load_B_nmajor((BPTR) + (size_t)32 * kE + (KK), kE, lane);                 \
  B3 = load_B_nmajor((BPTR) + (size_t)48 * kE + (KK), kE, lane);

#define GEMM_COMPUTE(A0, A1, B0, B1, B2, B3)                                     \
  c[0] = wmma_bf16(A0, B0, c[0]);  c[4] = wmma_bf16(A1, B0, c[4]);               \
  c[1] = wmma_bf16(A0, B1, c[1]);  c[5] = wmma_bf16(A1, B1, c[5]);               \
  c[2] = wmma_bf16(A0, B2, c[2]);  c[6] = wmma_bf16(A1, B2, c[6]);               \
  c[3] = wmma_bf16(A0, B3, c[3]);  c[7] = wmma_bf16(A1, B3, c[7]);

#define GEMM_BODY()                                                              \
  v16bf xA0, xA1, xB0, xB1, xB2, xB3;                                            \
  v16bf yA0, yA1, yB0, yB1, yB2, yB3;                                            \
  GEMM_LOAD(xA0, xA1, xB0, xB1, xB2, xB3, aptr, bptr, 0)                         \
  int kk = 0;                                                                    \
  for (; kk + 64 < kE; kk += 64) {                                               \
    GEMM_LOAD(yA0, yA1, yB0, yB1, yB2, yB3, aptr, bptr, kk + 32)                 \
    __builtin_prefetch(aptr + kk + 64, 0, 1);                                    \
    GEMM_COMPUTE(xA0, xA1, xB0, xB1, xB2, xB3)                                   \
    GEMM_LOAD(xA0, xA1, xB0, xB1, xB2, xB3, aptr, bptr, kk + 64)                 \
    GEMM_COMPUTE(yA0, yA1, yB0, yB1, yB2, yB3)                                   \
  }                                                                              \
  GEMM_LOAD(yA0, yA1, yB0, yB1, yB2, yB3, aptr, bptr, kk + 32)                   \
  GEMM_COMPUTE(xA0, xA1, xB0, xB1, xB2, xB3)                                     \
  GEMM_COMPUTE(yA0, yA1, yB0, yB1, yB2, yB3)

// ---------------- Kernel 1: QKV projection GEMM (32x64 tile/wave) ----------
// C[4096,3072] = xb @ WqkvT^T + bqkv; scatter Q,K -> [B,H,S,D], V -> [B,H,D,S]
__global__ __launch_bounds__(128) void mha_qkv_gemm(
    const __bf16* __restrict__ xb, const __bf16* __restrict__ WqkvT,
    const float* __restrict__ bqkv,
    __bf16* __restrict__ q, __bf16* __restrict__ k, __bf16* __restrict__ vt) {
  const int lane = threadIdx.x & 31, wave = threadIdx.x >> 5;
  const int m0 = blockIdx.y * 32;
  const int n0 = blockIdx.x * 256 + wave * 64;
  const __bf16* aptr = xb + (size_t)m0 * kE;
  const __bf16* bptr = WqkvT + (size_t)n0 * kE;

  v8f c[8];   // c[u*4+t]: u = 16-row sub-tile, t = 16-col sub-tile
#pragma unroll
  for (int i = 0; i < 8; ++i)
#pragma unroll
    for (int r = 0; r < 8; ++r) c[i][r] = 0.0f;

  GEMM_BODY()

  // Epilogue: constant-stride stores, part decode hoisted out of store loops.
  const int half = lane >> 4, n = lane & 15;
  const int bb = m0 >> 11;                        // batch (tile never straddles)
  const int s0 = (m0 & (kS - 1)) + half * 8;      // seq row of element r=0
#pragma unroll
  for (int t = 0; t < 4; ++t) {
    const int col  = n0 + t * 16 + n;
    const float bias = bqkv[col];
    const int part = col >> 10;                   // 0=q, 1=k, 2=v
    const int rem  = col & (kE - 1);
    const int h = rem >> 6, d = rem & (kD - 1);
    __bf16 vals[16];
#pragma unroll
    for (int u = 0; u < 2; ++u)
#pragma unroll
      for (int r = 0; r < 8; ++r) vals[u * 8 + r] = f32_to_bf16(c[u * 4 + t][r] + bias);
    if (part == 2) {
      __bf16* p = vt + ((size_t)(bb * kH + h) * kD + d) * kS + s0;
#pragma unroll
      for (int u = 0; u < 2; ++u)
#pragma unroll
        for (int r = 0; r < 8; ++r) p[u * 16 + r] = vals[u * 8 + r];
    } else {
      __bf16* p = (part ? k : q) + ((size_t)(bb * kH + h) * kS + s0) * kD + d;
#pragma unroll
      for (int u = 0; u < 2; ++u)
#pragma unroll
        for (int r = 0; r < 8; ++r) p[(u * 16 + r) * kD] = vals[u * 8 + r];
    }
  }
}

// ---------------- Kernel 2: causal flash attention ----------------
// One wave = 16 query rows; KV chunks of 32; online softmax; O = softmax(QK^T/8)V.
__global__ __launch_bounds__(128) void mha_flash_attn(
    const __bf16* __restrict__ q, const __bf16* __restrict__ k,
    const __bf16* __restrict__ vt, __bf16* __restrict__ attn) {
  __shared__ __bf16 lds_p[4][16 * 32];
  const int lane = threadIdx.x & 31, wave = threadIdx.x >> 5;
  const int bh = blockIdx.y;
  const int qb = blockIdx.x * 64 + wave * 16;
  const __bf16* qptr = q  + (size_t)bh * kS * kD;
  const __bf16* kptr = k  + (size_t)bh * kS * kD;
  const __bf16* vptr = vt + (size_t)bh * kD * kS;   // [D,S]
  const int half = lane >> 4, n = lane & 15;

  const v16bf aq0 = load_A_bf16(qptr + (size_t)qb * kD, kD, lane);       // D 0..31
  const v16bf aq1 = load_A_bf16(qptr + (size_t)qb * kD + 32, kD, lane);  // D 32..63

  v8f o[4];
  float mrow[8], lrow[8];
#pragma unroll
  for (int r = 0; r < 8; ++r) { mrow[r] = -3.0e38f; lrow[r] = 0.0f; }
#pragma unroll
  for (int t = 0; t < 4; ++t)
#pragma unroll
    for (int r = 0; r < 8; ++r) o[t][r] = 0.0f;

  const int nsteps = (qb + 16 + 31) >> 5;   // ceil((qb+16)/32) causal bound
  const float scale = 0.125f;               // 1/sqrt(64)
  const float log2e = 1.44269504f;

  for (int st = 0; st < nsteps; ++st) {
    const int kb = st * 32;
    // B(k=d, n=kv): K rows are contiguous per lane -> N-major loads on [S,D]
    v16bf b00 = load_B_nmajor(kptr + (size_t)kb * kD, kD, lane);
    v16bf b01 = load_B_nmajor(kptr + (size_t)kb * kD + 32, kD, lane);
    v16bf b10 = load_B_nmajor(kptr + (size_t)(kb + 16) * kD, kD, lane);
    v16bf b11 = load_B_nmajor(kptr + (size_t)(kb + 16) * kD + 32, kD, lane);
    // issue V loads now; consumed only after the softmax block below
    v16bf bv[4];
#pragma unroll
    for (int t = 0; t < 4; ++t)
      bv[t] = load_B_nmajor(vptr + (size_t)(t * 16) * kS + kb, kS, lane);

    v8f s0, s1;
#pragma unroll
    for (int r = 0; r < 8; ++r) { s0[r] = 0.0f; s1[r] = 0.0f; }
    s0 = wmma_bf16(aq0, b00, s0);
    s0 = wmma_bf16(aq1, b01, s0);
    s1 = wmma_bf16(aq0, b10, s1);
    s1 = wmma_bf16(aq1, b11, s1);

#pragma unroll
    for (int r = 0; r < 8; ++r) {
      const int qi = qb + r + half * 8;     // global query row for this element
      float x0 = s0[r] * scale;
      float x1 = s1[r] * scale;
      if (kb + n > qi)      x0 = -3.0e38f;  // causal mask
      if (kb + 16 + n > qi) x1 = -3.0e38f;
      float mx = fmaxf(x0, x1);             // row max across the 16-lane half
      mx = fmaxf(mx, __shfl_xor(mx, 1, 16));
      mx = fmaxf(mx, __shfl_xor(mx, 2, 16));
      mx = fmaxf(mx, __shfl_xor(mx, 4, 16));
      mx = fmaxf(mx, __shfl_xor(mx, 8, 16));
      const float mn  = fmaxf(mrow[r], mx);
      const float fac = exp2f((mrow[r] - mn) * log2e);
      const float p0  = exp2f((x0 - mn) * log2e);
      const float p1  = exp2f((x1 - mn) * log2e);
      float sum = p0 + p1;
      sum += __shfl_xor(sum, 1, 16);
      sum += __shfl_xor(sum, 2, 16);
      sum += __shfl_xor(sum, 4, 16);
      sum += __shfl_xor(sum, 8, 16);
      lrow[r] = lrow[r] * fac + sum;
      mrow[r] = mn;
#pragma unroll
      for (int t = 0; t < 4; ++t) o[t][r] *= fac;
      const int prow = r + half * 8;        // stage P tile (16x32) for A-frag reload
      lds_p[wave][prow * 32 + n]      = f32_to_bf16(p0);
      lds_p[wave][prow * 32 + 16 + n] = f32_to_bf16(p1);
    }
    const v16bf ap = load_A_bf16(&lds_p[wave][0], 32, lane);   // 2x ds_load_b128
#pragma unroll
    for (int t = 0; t < 4; ++t) o[t] = wmma_bf16(ap, bv[t], o[t]);
  }

  // Epilogue: one base pointer, constant immediate offsets.
  const int bb = bh >> 4, h = bh & 15;
  __bf16* p = attn + (size_t)(bb * kS + qb + half * 8) * kE + h * kD + n;
#pragma unroll
  for (int r = 0; r < 8; ++r) {
    const float inv = 1.0f / lrow[r];
#pragma unroll
    for (int t = 0; t < 4; ++t)
      p[(size_t)r * kE + t * 16] = f32_to_bf16(o[t][r] * inv);
  }
}

// ---------------- Kernel 3: output projection GEMM (32x64 tile/wave) -------
// out[4096,1024] = attn @ WoutT^T + bout (fp32 result)
__global__ __launch_bounds__(128) void mha_out_gemm(
    const __bf16* __restrict__ attn, const __bf16* __restrict__ WoutT,
    const float* __restrict__ bout, float* __restrict__ out) {
  const int lane = threadIdx.x & 31, wave = threadIdx.x >> 5;
  const int m0 = blockIdx.y * 32;
  const int n0 = blockIdx.x * 256 + wave * 64;
  const __bf16* aptr = attn + (size_t)m0 * kE;
  const __bf16* bptr = WoutT + (size_t)n0 * kE;

  v8f c[8];
#pragma unroll
  for (int i = 0; i < 8; ++i)
#pragma unroll
    for (int r = 0; r < 8; ++r) c[i][r] = 0.0f;

  GEMM_BODY()

  const int half = lane >> 4, n = lane & 15;
#pragma unroll
  for (int t = 0; t < 4; ++t) {
    const int col = n0 + t * 16 + n;
    const float bias = bout[col];
    float* p = out + (size_t)(m0 + half * 8) * kE + col;
#pragma unroll
    for (int u = 0; u < 2; ++u)
#pragma unroll
      for (int r = 0; r < 8; ++r)
        p[(size_t)(u * 16 + r) * kE] = c[u * 4 + t][r] + bias;
  }
}

extern "C" void kernel_launch(void* const* d_in, const int* in_sizes, int n_in,
                              void* d_out, int out_size, void* d_ws, size_t ws_size,
                              hipStream_t stream) {
  (void)in_sizes; (void)n_in; (void)out_size; (void)ws_size;
  const float* x    = (const float*)d_in[0];
  const float* Wqkv = (const float*)d_in[1];
  const float* bqkv = (const float*)d_in[2];
  const float* Wout = (const float*)d_in[3];
  const float* bout = (const float*)d_in[4];
  float* out = (float*)d_out;

  // Workspace layout (bf16 elements):
  //   xb[4096*1024]  WqkvT[3072*1024]  WoutT[1024*1024]
  //   q[B,H,S,D]  k[B,H,S,D]  vT[B,H,D,S]  attn[4096*1024]   (48 MB total)
  const size_t per = (size_t)kB * kH * kS * kD;
  __bf16* xb    = (__bf16*)d_ws;
  __bf16* wqkvT = xb + (size_t)kB * kS * kE;
  __bf16* woutT = wqkvT + (size_t)kE * k3E;
  __bf16* qbuf  = woutT + (size_t)kE * kE;
  __bf16* kbuf  = qbuf + per;
  __bf16* vtbuf = kbuf + per;
  __bf16* abuf  = vtbuf + per;

  const int nel_x = kB * kS * kE;
  cvt_f32_bf16<<<dim3((nel_x + 255) / 256), dim3(256), 0, stream>>>(x, xb, nel_x);
  transpose_f32_bf16<<<dim3(k3E / 32, kE / 32), dim3(256), 0, stream>>>(Wqkv, wqkvT, kE, k3E);
  transpose_f32_bf16<<<dim3(kE / 32, kE / 32), dim3(256), 0, stream>>>(Wout, woutT, kE, kE);

  dim3 blk(128);
  mha_qkv_gemm  <<<dim3(k3E / 256, (kB * kS) / 32), blk, 0, stream>>>(xb, wqkvT, bqkv, qbuf, kbuf, vtbuf);
  mha_flash_attn<<<dim3(kS / 64, kB * kH),          blk, 0, stream>>>(qbuf, kbuf, vtbuf, abuf);
  mha_out_gemm  <<<dim3(kE / 256, (kB * kS) / 32),  blk, 0, stream>>>(abuf, woutT, bout, out);
}
